// Custom_33990371181549
// MI455X (gfx1250) — compile-verified, matrix operand order checked by reading, not run on hardware
//
#include <hip/hip_runtime.h>

// ---------------------------------------------------------------------------
// CxWxW transposed attention for MI455X (gfx1250, wave32, WMMA).
//   b=16, c=8, h=w=512, heads=4, ch=2  ->  128 slices of 512x512x512 matmuls.
// Memory-bound (~1GB HBM traffic vs ~69 GFLOP); bf16 WMMA for the matmuls.
// ---------------------------------------------------------------------------

typedef __bf16 bf16_t;
typedef __attribute__((ext_vector_type(16))) __bf16          v16bf;
typedef __attribute__((ext_vector_type(8)))  float           v8f;
typedef __attribute__((ext_vector_type(16))) unsigned short  v16us;
typedef __attribute__((ext_vector_type(8)))  unsigned short  v8us;

#define B   16
#define C   8
#define HH  512
#define WW  512
#define C3  24

static __device__ __forceinline__ unsigned short f2bf(float f) {
    unsigned int u = __builtin_bit_cast(unsigned int, f);
    unsigned int r = (u + 0x7FFFu + ((u >> 16) & 1u)) >> 16;   // RNE
    return (unsigned short)r;
}
static __device__ __forceinline__ float bf2f(unsigned short s) {
    unsigned int u = ((unsigned int)s) << 16;
    return __builtin_bit_cast(float, u);
}
static __device__ __forceinline__ v8f zero8() {
    v8f z;
    #pragma unroll
    for (int i = 0; i < 8; ++i) z[i] = 0.f;
    return z;
}

// A/B fragment for v_wmma_f32_16x16x32_bf16.
// A (16xK=32): lane L<16 holds row L, k = {k0..k0+7, k0+16..k0+23};
//              lane L>=16 holds row L-16, k = {k0+8..k0+15, k0+24..k0+31}.
// B (K=32 x 16) mirrors this with lane <-> column; because we arrange the
// source matrices so the contraction dim is contiguous, BOTH fragments are
// two contiguous 16-byte loads per lane.
static __device__ __forceinline__ v16bf load_frag(const unsigned short* base,
                                                  int row0, int k0, int ld) {
    int lane = threadIdx.x & 31;
    int r = row0 + (lane & 15);
    int k = k0 + ((lane >> 4) << 3);
    const unsigned short* p = base + (size_t)r * ld + k;
    v8us lo = *(const v8us*)(p);
    v8us hi = *(const v8us*)(p + 16);
    union { v16us u; v16bf b; } cvt;
    #pragma unroll
    for (int i = 0; i < 8; ++i) { cvt.u[i] = lo[i]; cvt.u[8 + i] = hi[i]; }
    return cvt.b;
}

// ---------------------------------------------------------------------------
// K1: fused qkv = dwconv3x3(conv1x1(x)).  32x32 output tile + 1-px halo.
// Q,K written transposed [b][c][w][h] (h contiguous -> contraction-contiguous
// for S=QK^T).  V written natural [b][c][h][w] (w contiguous -> contraction-
// contiguous for O=P*V).
// ---------------------------------------------------------------------------
__global__ __launch_bounds__(256) void k1_qkv(
    const float* __restrict__ x, const float* __restrict__ w_qkv,
    const float* __restrict__ w_dw,
    unsigned short* __restrict__ Q, unsigned short* __restrict__ K,
    unsigned short* __restrict__ V)
{
    __shared__ float xs[C][34][36];
    __shared__ float ts[34][36];
    const int b   = blockIdx.z;
    const int tx0 = blockIdx.x * 32;   // w origin
    const int ty0 = blockIdx.y * 32;   // h origin
    const int tid = threadIdx.x;

    #pragma unroll
    for (int c = 0; c < C; ++c) {
        const float* xp = x + (((size_t)b * C + c) * HH) * WW;
        for (int i = tid; i < 34 * 34; i += 256) {
            int iy = i / 34, ix = i % 34;
            int gy = ty0 + iy - 1, gx = tx0 + ix - 1;
            float val = 0.f;
            if (gy >= 0 && gy < HH && gx >= 0 && gx < WW) val = xp[gy * WW + gx];
            xs[c][iy][ix] = val;
        }
    }
    __syncthreads();

    for (int o = 0; o < C3; ++o) {
        float wq[C];
        #pragma unroll
        for (int c = 0; c < C; ++c) wq[c] = w_qkv[o * C + c];
        for (int i = tid; i < 34 * 34; i += 256) {
            int iy = i / 34, ix = i % 34;
            float acc = 0.f;
            #pragma unroll
            for (int c = 0; c < C; ++c) acc += xs[c][iy][ix] * wq[c];
            ts[iy][ix] = acc;
        }
        __syncthreads();

        float wd[9];
        #pragma unroll
        for (int j = 0; j < 9; ++j) wd[j] = w_dw[o * 9 + j];
        const int t3 = o >> 3, c8 = o & 7;

        for (int i = tid; i < 32 * 32; i += 256) {
            int a0 = i >> 5, a1 = i & 31;
            // transposed outputs: make h fast-varying for coalesced stores
            int iy = (t3 < 2) ? a1 : a0;
            int ix = (t3 < 2) ? a0 : a1;
            float acc = 0.f;
            #pragma unroll
            for (int ky = 0; ky < 3; ++ky)
                #pragma unroll
                for (int kx = 0; kx < 3; ++kx)
                    acc += ts[iy + ky][ix + kx] * wd[ky * 3 + kx];
            int gy = ty0 + iy, gx = tx0 + ix;
            unsigned short bv = f2bf(acc);
            size_t ch = (size_t)b * C + c8;
            if (t3 == 0)      Q[(ch * WW + gx) * HH + gy] = bv;   // Qt [b][c][w][h]
            else if (t3 == 1) K[(ch * WW + gx) * HH + gy] = bv;   // Kt [b][c][w][h]
            else              V[(ch * HH + gy) * WW + gx] = bv;   // V  [b][c][h][w]
        }
        __syncthreads();
    }
}

// ---------------------------------------------------------------------------
// K2: L2-normalize each Q,K row (512 elems along h), one wave32 per row.
// ---------------------------------------------------------------------------
__global__ __launch_bounds__(256) void k2_norm(unsigned short* __restrict__ Q,
                                               unsigned short* __restrict__ K)
{
    const int gwave = (blockIdx.x * 256 + threadIdx.x) >> 5;
    const int lane  = threadIdx.x & 31;
    const int nrows = B * C * WW;                    // 65536 rows per tensor
    unsigned short* base =
        ((gwave < nrows) ? Q : K) + (size_t)(gwave % nrows) * HH;

    float vals[16];
    float ss = 0.f;
    #pragma unroll
    for (int i = 0; i < 16; ++i) {
        vals[i] = bf2f(base[lane + 32 * i]);
        ss += vals[i] * vals[i];
    }
    #pragma unroll
    for (int off = 16; off; off >>= 1) ss += __shfl_xor(ss, off, 32);
    float scale = 1.f / fmaxf(sqrtf(ss), 1e-12f);
    #pragma unroll
    for (int i = 0; i < 16; ++i) base[lane + 32 * i] = f2bf(vals[i] * scale);
}

// ---------------------------------------------------------------------------
// K3: S = (Q K^T) * temperature[head].  64x64 tile per WG, 4 waves, each wave
// a 16-row strip; 16 k-steps of 32 with v_wmma_f32_16x16x32_bf16.
// ---------------------------------------------------------------------------
__global__ __launch_bounds__(128) void k3_qk(
    const unsigned short* __restrict__ Q, const unsigned short* __restrict__ K,
    const float* __restrict__ temperature, float* __restrict__ S)
{
    const int s    = blockIdx.z;               // b*8 + channel
    const int mt   = blockIdx.y, nt = blockIdx.x;
    const int wid  = threadIdx.x >> 5;
    const int lane = threadIdx.x & 31;
    const unsigned short* qb = Q + (size_t)s * WW * HH;
    const unsigned short* kb = K + (size_t)s * WW * HH;
    const int m0 = mt * 64 + wid * 16;

    v8f acc[4];
    #pragma unroll
    for (int nn = 0; nn < 4; ++nn) acc[nn] = zero8();

    for (int kt = 0; kt < 16; ++kt) {
        const int k0 = kt * 32;
        v16bf a = load_frag(qb, m0, k0, HH);
        #pragma unroll
        for (int nn = 0; nn < 4; ++nn) {
            v16bf bb = load_frag(kb, nt * 64 + nn * 16, k0, HH);
            acc[nn] = __builtin_amdgcn_wmma_f32_16x16x32_bf16(
                false, a, false, bb, (short)0, acc[nn], false, false);
        }
    }

    const float temp = temperature[(s & 7) >> 1];   // head = channel/2
    float* sb = S + (size_t)s * WW * WW + (size_t)m0 * WW + nt * 64;
    const int mo = (lane >> 4) * 8, col = lane & 15;
    #pragma unroll
    for (int nn = 0; nn < 4; ++nn)
        #pragma unroll
        for (int j = 0; j < 8; ++j)
            sb[(size_t)(j + mo) * WW + nn * 16 + col] = acc[nn][j] * temp;
}

// ---------------------------------------------------------------------------
// K4: row softmax + O = P V.  32-row block per WG (256 thr = 8 waves).
// exp(S-max) staged as bf16 P~ in LDS (32 KB); 1/rowsum folded in epilogue.
// A-fragments come from LDS, B-fragments are contiguous loads of V (L2-res.).
// ---------------------------------------------------------------------------
__global__ __launch_bounds__(256) void k4_softmax_pv(
    const float* __restrict__ S, const unsigned short* __restrict__ V,
    float* __restrict__ O)
{
    __shared__ unsigned short pbuf[32 * 512];
    __shared__ float rinv[32];
    const int s  = blockIdx.y;
    const int rb = blockIdx.x;                 // rows rb*32 .. +32
    const int tid = threadIdx.x;

    // ---- softmax statistics (8 lanes per row) ----
    {
        const int row = tid >> 3, sub = tid & 7;
        const float* sr = S + (size_t)s * WW * WW + (size_t)(rb * 32 + row) * WW;
        float mx = -3.0e38f;
        for (int c = sub; c < WW; c += 8) mx = fmaxf(mx, sr[c]);
        #pragma unroll
        for (int off = 1; off < 8; off <<= 1) mx = fmaxf(mx, __shfl_xor(mx, off, 32));
        float sum = 0.f;
        for (int c = sub; c < WW; c += 8) {
            float p = __expf(sr[c] - mx);
            sum += p;
            pbuf[row * 512 + c] = f2bf(p);
        }
        #pragma unroll
        for (int off = 1; off < 8; off <<= 1) sum += __shfl_xor(sum, off, 32);
        if (sub == 0) rinv[row] = 1.f / sum;
    }
    __syncthreads();

    // ---- O = P~ * V via WMMA ----
    const int wid  = tid >> 5, lane = tid & 31;
    const int mblk = wid & 1;                  // which 16-row half of the block
    const int nq   = wid >> 1;                 // which 128-col quarter of h
    const unsigned short* vb = V + (size_t)s * HH * WW;
    float* ob = O + (size_t)s * WW * HH + (size_t)(rb * 32 + mblk * 16) * HH;
    const int m0 = mblk * 16;

    for (int nt = 0; nt < 8; ++nt) {
        const int n0 = nq * 128 + nt * 16;
        v8f acc = zero8();
        for (int kt = 0; kt < 16; ++kt) {
            const int k0 = kt * 32;
            v16bf a  = load_frag(pbuf, m0, k0, 512);     // LDS (ds_load_b128)
            v16bf bb = load_frag(vb,   n0, k0, WW);      // V[h=n][w=k] contiguous
            acc = __builtin_amdgcn_wmma_f32_16x16x32_bf16(
                false, a, false, bb, (short)0, acc, false, false);
        }
        const int mo = (lane >> 4) * 8, col = n0 + (lane & 15);
        #pragma unroll
        for (int j = 0; j < 8; ++j) {
            float sc = rinv[m0 + mo + j];
            ob[(size_t)(mo + j) * HH + col] = acc[j] * sc;
        }
    }
}

// ---------------------------------------------------------------------------
// K5: transpose back (O is [b][c][w][h]) + 1x1 output projection (8x8).
// ---------------------------------------------------------------------------
__global__ __launch_bounds__(256) void k5_proj(
    const float* __restrict__ O, const float* __restrict__ w_out,
    float* __restrict__ out)
{
    __shared__ float ot[C][32][33];
    const int b   = blockIdx.z;
    const int wx0 = blockIdx.x * 32, hy0 = blockIdx.y * 32;
    const int tid = threadIdx.x;

    #pragma unroll
    for (int c = 0; c < C; ++c) {
        const float* op = O + (((size_t)b * C + c) * WW + wx0) * HH + hy0;
        for (int i = tid; i < 1024; i += 256) {
            int iw = i >> 5, ih = i & 31;
            ot[c][iw][ih] = op[(size_t)iw * HH + ih];   // h contiguous
        }
    }
    __syncthreads();

    float wo[64];
    #pragma unroll
    for (int i = 0; i < 64; ++i) wo[i] = w_out[i];      // uniform -> SGPRs

    for (int i = tid; i < 1024; i += 256) {
        int ih = i >> 5, iw = i & 31;                    // w fast -> coalesced out
        float vals[C];
        #pragma unroll
        for (int c = 0; c < C; ++c) vals[c] = ot[c][iw][ih];
        #pragma unroll
        for (int o = 0; o < C; ++o) {
            float acc = 0.f;
            #pragma unroll
            for (int c = 0; c < C; ++c) acc += wo[o * C + c] * vals[c];
            out[(((size_t)b * C + o) * HH + (hy0 + ih)) * WW + (wx0 + iw)] = acc;
        }
    }
}

// ---------------------------------------------------------------------------
extern "C" void kernel_launch(void* const* d_in, const int* in_sizes, int n_in,
                              void* d_out, int out_size, void* d_ws, size_t ws_size,
                              hipStream_t stream)
{
    const float* x      = (const float*)d_in[0];
    const float* w_qkv  = (const float*)d_in[1];
    const float* w_dw   = (const float*)d_in[2];
    const float* w_out  = (const float*)d_in[3];
    const float* temp   = (const float*)d_in[4];
    float* out = (float*)d_out;

    char* ws = (char*)d_ws;
    const size_t TBYTES = (size_t)B * C * HH * WW * 2;      // 64 MB per bf16 tensor
    unsigned short* Q = (unsigned short*)(ws);
    unsigned short* K = (unsigned short*)(ws + TBYTES);
    unsigned short* V = (unsigned short*)(ws + 2 * TBYTES);
    float* S = (float*)(ws + 3 * TBYTES);                   // 128 MB
    float* O = (float*)(ws + 3 * TBYTES + (size_t)128 * WW * WW * 4);

    k1_qkv<<<dim3(WW / 32, HH / 32, B), dim3(256), 0, stream>>>(
        x, w_qkv, w_dw, Q, K, V);

    // 2 tensors * 65536 rows, 1 wave/row, 8 waves/block -> 16384 blocks
    k2_norm<<<dim3(16384), dim3(256), 0, stream>>>(Q, K);

    k3_qk<<<dim3(WW / 64, WW / 64, B * C), dim3(128), 0, stream>>>(
        Q, K, temp, S);

    k4_softmax_pv<<<dim3(WW / 32, B * C), dim3(256), 0, stream>>>(S, V, O);

    k5_proj<<<dim3(WW / 32, HH / 32, B), dim3(256), 0, stream>>>(O, w_out, out);
}
